// EasyPreTrainLatentAttentionWithIndexer_69681549410472
// MI455X (gfx1250) — matrix operand order repaired and checked by reading, not compiled, not run on hardware
//
#include <hip/hip_runtime.h>

#define DIMX   2048
#define NHX    16
#define QLORA  1536
#define KVLORA 512
#define NOPEX  128
#define ROPEX  64
#define VDIMX  128
#define QKX    192
#define BBX    2
#define SSX    2048
#define NTOK   (BBX*SSX)
#define EPSX   1.1920929e-07f

typedef __attribute__((ext_vector_type(16))) __bf16 v16bf;
typedef __attribute__((ext_vector_type(8)))  __bf16 v8bf;
typedef __attribute__((ext_vector_type(8)))  float  v8f;

__device__ __forceinline__ unsigned short f2bf(float f) {
  unsigned int u = __float_as_uint(f);
  u += 0x7FFFu + ((u >> 16) & 1u);
  return (unsigned short)(u >> 16);
}

__device__ __forceinline__ v16bf cat8(v8bf lo, v8bf hi) {
  return __builtin_shufflevector(lo, hi, 0,1,2,3,4,5,6,7,8,9,10,11,12,13,14,15);
}

// LDS byte offset of a generic pointer to __shared__ (low 32 bits hold the
// AS3 offset after the addrspacecast to flat).
__device__ __forceinline__ unsigned ldsOff(const void* p) {
  return (unsigned)(unsigned long long)p;
}

// CDNA5 async global->LDS copy, 16B per lane, GVS addressing
// (SGPR64 base + per-lane VGPR byte offset). Tracked by ASYNCcnt.
__device__ __forceinline__ void async_b128(unsigned lds, const void* base,
                                           unsigned byteoff) {
  asm volatile("global_load_async_to_lds_b128 %0, %1, %2"
               :: "v"(lds), "v"(byteoff), "s"(base)
               : "memory");
}
__device__ __forceinline__ void wait_async_le10() {
  asm volatile("s_wait_asynccnt 0xa" ::: "memory");
}
__device__ __forceinline__ void wait_async_le5() {
  asm volatile("s_wait_asynccnt 0x5" ::: "memory");
}
__device__ __forceinline__ void wait_async_0() {
  asm volatile("s_wait_asynccnt 0x0" ::: "memory");
}

// ---------------- f32 -> bf16 convert (grid-stride) ----------------
__global__ void cvt_f32_bf16(const float* __restrict__ in,
                             unsigned short* __restrict__ out, long n) {
  long i = (long)blockIdx.x * blockDim.x + threadIdx.x;
  long stride = (long)gridDim.x * blockDim.x;
  for (; i < n; i += stride) out[i] = f2bf(in[i]);
}

// ---------------- GEMM: C[M,N] = A[M,K] * W[N,K]^T + bias ----------------
// A, W bf16 row-major (contiguous along K); C f32.
// Block 256 threads = 8 waves (2x4), block tile 64x256, wave tile 32x64,
// K-step 32. Quad-buffered LDS fed by async-to-LDS with 2 tiles in flight,
// branch-free steady-state loop, 8 WMMAs per barrier.
__launch_bounds__(256)
__global__ void gemm_bf16_wmma(const unsigned short* __restrict__ A,
                               const unsigned short* __restrict__ W,
                               const float* __restrict__ bias,
                               float* __restrict__ C,
                               int M, int N, int K) {
  __shared__ unsigned short As[4][64][32];
  __shared__ unsigned short Bs[4][256][32];
  const int tid  = threadIdx.x;
  const int lane = tid & 31;
  const int wave = tid >> 5;      // 0..7
  const int wm   = wave >> 2;     // 0..1
  const int wn   = wave & 3;      // 0..3
  const int m0   = blockIdx.y * 64;
  const int n0   = blockIdx.x * 256;

  // cooperative tile load mapping
  const int arow = tid >> 2;           // 0..63
  const int acol = (tid & 3) * 8;      // 0,8,16,24
  const long agrow = m0 + arow;
  long bgrow = n0 + tid;               // one full B row per thread
  if (bgrow > N - 1) bgrow = N - 1;    // clamp (N=576 case)

  // fragment addressing
  const int ra0  = wm * 32 + (lane & 15);
  const int base = (lane >> 4) * 16;
  const int cb0  = wn * 64 + (lane & 15);
  const int klo  = (lane >> 4) * 16;

  v8f acc[2][4];
#pragma unroll
  for (int i = 0; i < 2; ++i)
#pragma unroll
    for (int j = 0; j < 4; ++j) acc[i][j] = (v8f){};

  auto issue = [&](int k0, int buf) {
    async_b128(ldsOff(&As[buf][arow][acol]), A,
               (unsigned)((agrow * K + k0 + acol) * 2));
#pragma unroll
    for (int q = 0; q < 4; ++q)
      async_b128(ldsOff(&Bs[buf][tid][q * 8]), W,
                 (unsigned)((bgrow * K + k0 + q * 8) * 2));
  };

  auto compute = [&](int buf) {
    v16bf a0 = *(const v16bf*)&As[buf][ra0][base];
    v16bf a1 = *(const v16bf*)&As[buf][ra0 + 16][base];
#pragma unroll
    for (int j = 0; j < 4; ++j) {
      v16bf b = *(const v16bf*)&Bs[buf][cb0 + j * 16][klo];
      acc[0][j] = __builtin_amdgcn_wmma_f32_16x16x32_bf16(
          false, a0, false, b, (short)0, acc[0][j], false, false);
      acc[1][j] = __builtin_amdgcn_wmma_f32_16x16x32_bf16(
          false, a1, false, b, (short)0, acc[1][j], false, false);
    }
  };

  const int nk = K >> 5;   // >= 16 for all our shapes
  issue(0, 0);
  issue(32, 1);
  int k = 0;
  for (; k + 2 < nk; ++k) {            // branch-free steady state
    issue((k + 2) * 32, (k + 2) & 3);
    wait_async_le10();                 // tile k landed; k+1,k+2 in flight
    __syncthreads();
    compute(k & 3);
  }
  wait_async_le5();                    // tile nk-2 landed
  __syncthreads();
  compute(k & 3);
  ++k;
  wait_async_0();                      // tile nk-1 landed
  __syncthreads();
  compute(k & 3);

  // epilogue: bias + store (guard columns for N not multiple of 256)
#pragma unroll
  for (int j = 0; j < 4; ++j) {
    const int n = n0 + wn * 64 + j * 16 + (lane & 15);
    if (n >= N) continue;
    const float bn = bias[n];
#pragma unroll
    for (int i = 0; i < 2; ++i) {
      const int mb = m0 + wm * 32 + i * 16 + (lane >> 4) * 8;
#pragma unroll
      for (int t = 0; t < 8; ++t)
        C[(long)(mb + t) * N + n] = acc[i][j][t] + bn;
    }
  }
}

// ---------------- RMSNorm (one row per block), f32 in -> bf16 out ----------------
__launch_bounds__(256)
__global__ void rmsnorm_bf16(const float* __restrict__ in, int in_stride, int len,
                             const float* __restrict__ w,
                             unsigned short* __restrict__ out, int out_stride) {
  __shared__ float red[256];
  const long row = blockIdx.x;
  const float* x = in + row * in_stride;
  float s = 0.0f;
  for (int i = threadIdx.x; i < len; i += 256) { float v = x[i]; s += v * v; }
  red[threadIdx.x] = s;
  __syncthreads();
  for (int off = 128; off > 0; off >>= 1) {
    if ((int)threadIdx.x < off) red[threadIdx.x] += red[threadIdx.x + off];
    __syncthreads();
  }
  const float inv = rsqrtf(red[0] / (float)len + EPSX);
  unsigned short* o = out + row * out_stride;
  for (int i = threadIdx.x; i < len; i += 256) o[i] = f2bf(x[i] * inv * w[i]);
}

// ---------------- Assemble Q: [B,S,NH,QK] f32 + rope -> [B,NH,S,QK] bf16 ----------------
__global__ void assemble_q(const float* __restrict__ Qf,
                           const float* __restrict__ freqs,
                           unsigned short* __restrict__ Qbf) {
  long idx = (long)blockIdx.x * blockDim.x + threadIdx.x;
  const long total = (long)BBX * NHX * SSX * QKX;
  if (idx >= total) return;
  int d = (int)(idx % QKX); long t = idx / QKX;
  int s = (int)(t % SSX); t /= SSX;
  int h = (int)(t % NHX); int b = (int)(t / NHX);
  const float* qrow = Qf + ((long)(b * SSX + s) * NHX + h) * QKX;
  float val;
  if (d < NOPEX) {
    val = qrow[d];
  } else {
    int tt = d - NOPEX, p = tt >> 1;
    float f = freqs[(long)s * (ROPEX / 2) + p];
    float c = cosf(f), sn = sinf(f);
    float x0 = qrow[NOPEX + 2 * p], x1 = qrow[NOPEX + 2 * p + 1];
    val = (tt & 1) ? (x0 * sn + x1 * c) : (x0 * c - x1 * sn);
  }
  Qbf[idx] = f2bf(val);
}

// ---------------- Assemble K: nope from kv, rope (shared) from kv_kr -> [B,NH,S,QK] bf16 ----------------
__global__ void assemble_k(const float* __restrict__ kvf,
                           const float* __restrict__ kv_kr,
                           const float* __restrict__ freqs,
                           unsigned short* __restrict__ Kbf) {
  long idx = (long)blockIdx.x * blockDim.x + threadIdx.x;
  const long total = (long)BBX * NHX * SSX * QKX;
  if (idx >= total) return;
  int d = (int)(idx % QKX); long t = idx / QKX;
  int s = (int)(t % SSX); t /= SSX;
  int h = (int)(t % NHX); int b = (int)(t / NHX);
  float val;
  if (d < NOPEX) {
    val = kvf[((long)(b * SSX + s) * NHX + h) * (NOPEX + VDIMX) + d];
  } else {
    int tt = d - NOPEX, p = tt >> 1;
    float f = freqs[(long)s * (ROPEX / 2) + p];
    float c = cosf(f), sn = sinf(f);
    const float* kr = kv_kr + (long)(b * SSX + s) * (KVLORA + ROPEX) + KVLORA;
    float x0 = kr[2 * p], x1 = kr[2 * p + 1];
    val = (tt & 1) ? (x0 * sn + x1 * c) : (x0 * c - x1 * sn);
  }
  Kbf[idx] = f2bf(val);
}

// ---------------- Assemble V transposed: kv[...,128:256] -> [B,NH,VDIM,S] bf16 ----------------
__global__ void assemble_v(const float* __restrict__ kvf,
                           unsigned short* __restrict__ Vt) {
  long idx = (long)blockIdx.x * blockDim.x + threadIdx.x;
  const long total = (long)BBX * NHX * VDIMX * SSX;
  if (idx >= total) return;
  int s = (int)(idx % SSX); long t = idx / SSX;
  int d = (int)(t % VDIMX); t /= VDIMX;
  int h = (int)(t % NHX); int b = (int)(t / NHX);
  Vt[idx] = f2bf(kvf[((long)(b * SSX + s) * NHX + h) * (NOPEX + VDIMX) + NOPEX + d]);
}

// ---------------- Flash attention: 1 wave per 16-row Q tile, 32 keys/step ----------------
// Q [B,NH,S,192] bf16, K [B,NH,S,192] bf16, V^T [B,NH,128,S] bf16
// out: attn bf16 [B,S,NH*128]
__launch_bounds__(128)
__global__ void flash_attn(const unsigned short* __restrict__ Qbf,
                           const unsigned short* __restrict__ Kbf,
                           const unsigned short* __restrict__ Vt,
                           unsigned short* __restrict__ Obf) {
  __shared__ unsigned short Pl[4][16][32];
  const int lane = threadIdx.x & 31;
  const int wave = threadIdx.x >> 5;
  const int T    = blockIdx.x * 4 + wave;
  const int qt   = T & (SSX / 16 - 1);   // 0..127
  const int bh   = T >> 7;               // 0..31
  const int half = lane >> 4;
  const int ln   = lane & 15;

  const unsigned short* Qbase = Qbf + ((long)bh * SSX + qt * 16) * QKX;
  v16bf qf[6];
  {
    const int base = half * 16;
#pragma unroll
    for (int sl = 0; sl < 6; ++sl) {
      const unsigned short* p = Qbase + (long)ln * QKX + sl * 32;
      qf[sl] = *(const v16bf*)(p + base);
    }
  }

  v8f o[8];
#pragma unroll
  for (int n = 0; n < 8; ++n) o[n] = (v8f){};
  float rmax[8], rsum[8];
#pragma unroll
  for (int i = 0; i < 8; ++i) { rmax[i] = -1e30f; rsum[i] = 0.0f; }

  const float scale = 0.07216878364870323f;  // 1/sqrt(192)
  const int nkb = (qt * 16 + 16 + 31) >> 5;

  for (int it = 0; it < nkb; ++it) {
    const int kb = it * 32;
    v8f s0 = {}, s1 = {};
    const unsigned short* Kbase = Kbf + ((long)bh * SSX + kb) * QKX;
    const int klo = half * 16;
#pragma unroll
    for (int sl = 0; sl < 6; ++sl) {
      v16bf b0 = *(const v16bf*)(Kbase + (long)ln * QKX + sl * 32 + klo);
      v16bf b1 = *(const v16bf*)(Kbase + (long)(16 + ln) * QKX + sl * 32 + klo);
      s0 = __builtin_amdgcn_wmma_f32_16x16x32_bf16(false, qf[sl], false, b0,
                                                   (short)0, s0, false, false);
      s1 = __builtin_amdgcn_wmma_f32_16x16x32_bf16(false, qf[sl], false, b1,
                                                   (short)0, s1, false, false);
    }

    const int key0 = kb + ln, key1 = kb + 16 + ln;
    float p0[8], p1[8], alpha[8];
#pragma unroll
    for (int i = 0; i < 8; ++i) {
      const int q = qt * 16 + i + half * 8;
      float v0 = (key0 <= q) ? s0[i] * scale : -1e30f;
      float v1 = (key1 <= q) ? s1[i] * scale : -1e30f;
      float m = fmaxf(v0, v1);
#pragma unroll
      for (int off = 1; off < 16; off <<= 1) m = fmaxf(m, __shfl_xor(m, off, 32));
      const float mn = fmaxf(rmax[i], m);
      alpha[i] = __expf(rmax[i] - mn);
      rmax[i] = mn;
      float e0 = __expf(v0 - mn), e1 = __expf(v1 - mn);
      float ps = e0 + e1;
#pragma unroll
      for (int off = 1; off < 16; off <<= 1) ps += __shfl_xor(ps, off, 32);
      rsum[i] = rsum[i] * alpha[i] + ps;
      p0[i] = e0; p1[i] = e1;
    }
#pragma unroll
    for (int n = 0; n < 8; ++n)
#pragma unroll
      for (int i = 0; i < 8; ++i) o[n][i] *= alpha[i];

#pragma unroll
    for (int i = 0; i < 8; ++i) {
      Pl[wave][i + half * 8][ln]      = f2bf(p0[i]);
      Pl[wave][i + half * 8][ln + 16] = f2bf(p1[i]);
    }
    v16bf pf;
    {
      const int base = half * 16;
      pf = *(const v16bf*)&Pl[wave][ln][base];
    }

    const unsigned short* Vb = Vt + (long)bh * VDIMX * SSX + kb + half * 16;
#pragma unroll
    for (int n = 0; n < 8; ++n) {
      v16bf vf = *(const v16bf*)(Vb + (long)(n * 16 + ln) * SSX);
      o[n] = __builtin_amdgcn_wmma_f32_16x16x32_bf16(false, pf, false, vf,
                                                     (short)0, o[n], false, false);
    }
  }

  const int b = bh >> 4, h = bh & 15;
#pragma unroll
  for (int n = 0; n < 8; ++n)
#pragma unroll
    for (int i = 0; i < 8; ++i) {
      const int q = qt * 16 + i + half * 8;
      Obf[(long)(b * SSX + q) * (NHX * VDIMX) + h * VDIMX + n * 16 + ln] =
          f2bf(o[n][i] / rsum[i]);
    }
}

// ---------------- host-side launch ----------------
extern "C" void kernel_launch(void* const* d_in, const int* in_sizes, int n_in,
                              void* d_out, int out_size, void* d_ws, size_t ws_size,
                              hipStream_t stream) {
  const float* x        = (const float*)d_in[0];
  const float* freqs    = (const float*)d_in[1];
  // d_in[2] = causal mask (applied analytically inside flash_attn)
  const float* Wq_down  = (const float*)d_in[3];
  const float* Wq_down_b= (const float*)d_in[4];
  const float* q_norm_w = (const float*)d_in[5];
  const float* Wq_up    = (const float*)d_in[6];
  const float* Wq_up_b  = (const float*)d_in[7];
  const float* Wkv_down = (const float*)d_in[8];
  const float* Wkv_down_b=(const float*)d_in[9];
  const float* kv_norm_w= (const float*)d_in[10];
  const float* Wkv_up   = (const float*)d_in[11];
  const float* Wkv_up_b = (const float*)d_in[12];
  const float* Wo       = (const float*)d_in[13];
  const float* Wo_b     = (const float*)d_in[14];
  float* out = (float*)d_out;

  char* ws = (char*)d_ws;
  size_t off = 0;
  auto alloc = [&](size_t bytes) -> void* {
    void* p = ws + off;
    off = (off + bytes + 255) & ~(size_t)255;
    return p;
  };

  unsigned short* x_bf      = (unsigned short*)alloc((size_t)NTOK * DIMX * 2);
  unsigned short* wq_down_bf= (unsigned short*)alloc((size_t)QLORA * DIMX * 2);
  unsigned short* wq_up_bf  = (unsigned short*)alloc((size_t)NHX * QKX * QLORA * 2);
  unsigned short* wkv_down_bf=(unsigned short*)alloc((size_t)(KVLORA + ROPEX) * DIMX * 2);
  unsigned short* wkv_up_bf = (unsigned short*)alloc((size_t)NHX * (NOPEX + VDIMX) * KVLORA * 2);
  unsigned short* wo_bf     = (unsigned short*)alloc((size_t)DIMX * NHX * VDIMX * 2);
  float* q_lat_raw          = (float*)alloc((size_t)NTOK * QLORA * 4);
  unsigned short* q_lat_bf  = (unsigned short*)alloc((size_t)NTOK * QLORA * 2);
  float* Qf32               = (float*)alloc((size_t)NTOK * NHX * QKX * 4);
  float* kv_kr              = (float*)alloc((size_t)NTOK * (KVLORA + ROPEX) * 4);
  unsigned short* kv_lat_bf = (unsigned short*)alloc((size_t)NTOK * KVLORA * 2);
  float* kvf32              = (float*)alloc((size_t)NTOK * NHX * (NOPEX + VDIMX) * 4);
  unsigned short* Qbf       = (unsigned short*)alloc((size_t)BBX * NHX * SSX * QKX * 2);
  unsigned short* Kbf       = (unsigned short*)alloc((size_t)BBX * NHX * SSX * QKX * 2);
  unsigned short* Vt        = (unsigned short*)alloc((size_t)BBX * NHX * VDIMX * SSX * 2);
  unsigned short* attn_bf   = (unsigned short*)alloc((size_t)NTOK * NHX * VDIMX * 2);

  auto cvt = [&](const float* src, unsigned short* dst, long n) {
    cvt_f32_bf16<<<2048, 256, 0, stream>>>(src, dst, n);
  };

  cvt(x, x_bf, (long)NTOK * DIMX);
  cvt(Wq_down, wq_down_bf, (long)QLORA * DIMX);
  cvt(Wq_up, wq_up_bf, (long)NHX * QKX * QLORA);
  cvt(Wkv_down, wkv_down_bf, (long)(KVLORA + ROPEX) * DIMX);
  cvt(Wkv_up, wkv_up_bf, (long)NHX * (NOPEX + VDIMX) * KVLORA);
  cvt(Wo, wo_bf, (long)DIMX * NHX * VDIMX);

  // q path: down-proj, rmsnorm, up-proj
  gemm_bf16_wmma<<<dim3((QLORA + 255) / 256, NTOK / 64), 256, 0, stream>>>(
      x_bf, wq_down_bf, Wq_down_b, q_lat_raw, NTOK, QLORA, DIMX);
  rmsnorm_bf16<<<NTOK, 256, 0, stream>>>(q_lat_raw, QLORA, QLORA, q_norm_w,
                                         q_lat_bf, QLORA);
  gemm_bf16_wmma<<<dim3((NHX * QKX + 255) / 256, NTOK / 64), 256, 0, stream>>>(
      q_lat_bf, wq_up_bf, Wq_up_b, Qf32, NTOK, NHX * QKX, QLORA);

  // kv path: down-proj (N=576, guarded tiles), rmsnorm on first 512, up-proj
  gemm_bf16_wmma<<<dim3((KVLORA + ROPEX + 255) / 256, NTOK / 64), 256, 0, stream>>>(
      x_bf, wkv_down_bf, Wkv_down_b, kv_kr, NTOK, KVLORA + ROPEX, DIMX);
  rmsnorm_bf16<<<NTOK, 256, 0, stream>>>(kv_kr, KVLORA + ROPEX, KVLORA, kv_norm_w,
                                         kv_lat_bf, KVLORA);
  gemm_bf16_wmma<<<dim3((NHX * (NOPEX + VDIMX) + 255) / 256, NTOK / 64), 256, 0, stream>>>(
      kv_lat_bf, wkv_up_bf, Wkv_up_b, kvf32, NTOK, NHX * (NOPEX + VDIMX), KVLORA);

  // assemble Q/K (with rope) and V^T in attention-friendly layouts
  {
    long tq = (long)BBX * NHX * SSX * QKX;
    assemble_q<<<(unsigned)((tq + 255) / 256), 256, 0, stream>>>(Qf32, freqs, Qbf);
    assemble_k<<<(unsigned)((tq + 255) / 256), 256, 0, stream>>>(kvf32, kv_kr, freqs, Kbf);
    long tv = (long)BBX * NHX * VDIMX * SSX;
    assemble_v<<<(unsigned)((tv + 255) / 256), 256, 0, stream>>>(kvf32, Vt);
  }

  // flash attention: B*NH*(S/16) tiles, 4 waves per block
  flash_attn<<<(BBX * NHX * (SSX / 16)) / 4, 128, 0, stream>>>(Qbf, Kbf, Vt, attn_bf);

  // output projection
  gemm_bf16_wmma<<<dim3((DIMX + 255) / 256, NTOK / 64), 256, 0, stream>>>(
      attn_bf, wo_bf, Wo_b, out, NTOK, DIMX, NHX * VDIMX);

  (void)in_sizes; (void)n_in; (void)out_size; (void)ws_size;
}
